// HybridQueryGNN_46806553592345
// MI455X (gfx1250) — matrix operand order repaired and checked by reading, not compiled, not run on hardware
//
#include <hip/hip_runtime.h>
#include <hip/hip_bf16.h>

typedef __attribute__((ext_vector_type(16))) _Float16 v16h;
typedef __attribute__((ext_vector_type(8)))  float    v8f;

#define NN 100000
#define EE 1600000
#define ET (EE + NN)
#define GG 1000

// ---------------- Tensor Data Mover (gfx1250) ------------------------------
#if defined(__gfx1250__) && __has_builtin(__builtin_amdgcn_tensor_load_to_lds)
#define USE_TDM 1
#endif

#ifdef USE_TDM
typedef __attribute__((ext_vector_type(4))) unsigned int tdm_v4u;
typedef __attribute__((ext_vector_type(8))) int          tdm_v8i;
typedef __attribute__((ext_vector_type(4))) int          tdm_v4i;

// 1-D contiguous tile load: nelem f32 elements from gptr -> LDS byte offset.
__device__ __forceinline__ void tdm_load_1d(unsigned lds_off, const void* gptr,
                                            unsigned nelem) {
    unsigned long long ga = (unsigned long long)gptr;
    tdm_v4u g0;
    g0[0] = 1u;                                   // count=1 (valid user D#)
    g0[1] = lds_off;                              // lds_addr (bytes)
    g0[2] = (unsigned)(ga & 0xFFFFFFFFull);       // global_addr[31:0]
    g0[3] = (unsigned)((ga >> 32) & 0x01FFFFFFull) | (2u << 30);  // [56:32] + type=2
    tdm_v8i g1;
    g1[0] = (int)(2u << 16);                      // data_size=2 (4 bytes)
    g1[1] = (int)((nelem & 0xFFFFu) << 16);       // tensor_dim0[15:0]
    g1[2] = (int)((nelem >> 16) | (1u << 16));    // tensor_dim0[31:16], tensor_dim1=1
    g1[3] = (int)((nelem & 0xFFFFu) << 16);       // tile_dim0 = nelem
    g1[4] = 1;                                    // tile_dim1 = 1
    g1[5] = (int)nelem;                           // tensor_dim0_stride
    g1[6] = 0;
    g1[7] = 0;
    tdm_v4i gz = {0, 0, 0, 0};
#if __clang_major__ >= 23
    tdm_v8i gz8 = {0, 0, 0, 0, 0, 0, 0, 0};
    __builtin_amdgcn_tensor_load_to_lds(g0, g1, gz, gz, gz8, 0);
#else
    __builtin_amdgcn_tensor_load_to_lds(g0, g1, gz, gz, 0);
#endif
}
#endif

// ---------- order-preserving float <-> uint encoding for atomicMax ----------
__device__ __forceinline__ unsigned fenc(float f) {
    unsigned u = __float_as_uint(f);
    return (u & 0x80000000u) ? ~u : (u | 0x80000000u);
}
__device__ __forceinline__ float fdec(unsigned e) {
    unsigned u = (e & 0x80000000u) ? (e & 0x7FFFFFFFu) : ~e;
    return __uint_as_float(u);
}

__device__ __forceinline__ float wave_sum(float v) {
    #pragma unroll
    for (int m = 16; m >= 1; m >>= 1) v += __shfl_xor(v, m, 32);
    return v;
}

// ---------- zero init of accumulator region ----------
__global__ void zero_ws(float* p, size_t n) {
    size_t i = (size_t)blockIdx.x * blockDim.x + threadIdx.x;
    size_t stride = (size_t)gridDim.x * blockDim.x;
    for (; i < n; i += stride) p[i] = 0.0f;
}

// ---------- layer-1 node transform: xl1 = xcat @ Wl1, xr1 = xcat @ Wr1 ------
// Stage the raw contiguous 16x26 row block (1664B) into LDS via TDM, then
// apply the col-24 skip / K-padding while building the A fragment.
// Grid: 6250 M-tiles; block 256 = 8 waves; wave w: matrix = w>>2, ntile = w&3.
__global__ void transform1(const float* __restrict__ x,
                           const float* __restrict__ Wl,
                           const float* __restrict__ Wr,
                           float* __restrict__ xl, float* __restrict__ xr) {
    __shared__ float xs[16 * 26];
    int mtile = blockIdx.x;
    int tid = threadIdx.x;
    const float* src = x + (size_t)mtile * 16 * 26;

#ifdef USE_TDM
    if (tid < 32) {
        tdm_load_1d((unsigned)(size_t)(void*)xs, src, 16 * 26);
        __builtin_amdgcn_s_wait_tensorcnt(0);
    }
    __syncthreads();
#else
    for (int i = tid; i < 16 * 26; i += 256) xs[i] = src[i];
    __syncthreads();
#endif

    int wave = tid >> 5;
    int lane = tid & 31;
    int wsel = wave >> 2;          // 0 = Wl, 1 = Wr
    int ntile = wave & 3;          // 4 tiles of 16 cols -> 64 outputs
    const float* W = wsel ? Wr : Wl;
    float* Y = wsel ? xr : xl;

    int r = lane & 15;
    int half = lane >> 4;
    int coln = ntile * 16 + r;     // B/C/D: lane holds column N=r

    v16h a, b;
    v8f c = {};

    // A 16x32 f16 layout (ISA 7.12.2): VGPR p holds a K pair for row M=r
    #pragma unroll
    for (int p = 0; p < 8; p++) {
        int k0 = ((p >> 2) << 4) + half * 8 + ((p & 3) << 1);
        #pragma unroll
        for (int q = 0; q < 2; q++) {
            int k = k0 + q;
            float v = 0.0f;
            if (k < 25) v = xs[r * 26 + ((k < 24) ? k : 25)];
            a[2 * p + q] = (_Float16)v;
        }
    }
    // B 32x16 f16: same K order as A (VGPR v holds K pair 16*(v>>2)+8*half+2*(v&3))
    #pragma unroll
    for (int v = 0; v < 8; v++) {
        int kk = ((v >> 2) << 4) + half * 8 + ((v & 3) << 1);
        b[2 * v]     = (_Float16)((kk     < 25) ? W[(size_t)kk * 64 + coln]       : 0.0f);
        b[2 * v + 1] = (_Float16)((kk + 1 < 25) ? W[(size_t)(kk + 1) * 64 + coln] : 0.0f);
    }

    c = __builtin_amdgcn_wmma_f32_16x16x32_f16(false, a, false, b,
                                               (short)0, c, false, false);

    // C/D 16x16 f32: VGPR v -> M = v + 8*half, N = coln
    #pragma unroll
    for (int v = 0; v < 8; v++) {
        Y[(size_t)(mtile * 16 + v + 8 * half) * 64 + coln] = c[v];
    }
}

// ---------- layer-2 node transform: xl2/xr2 = h1 @ Wl2/Wr2 (64 -> 32) -------
// Grid: 6250 M-tiles; block 128 = 4 waves; wave w: matrix = w>>1, ntile = w&1.
// A tile (16x64, contiguous 4KB) staged into LDS via TDM.
__global__ void transform2(const float* __restrict__ h1,
                           const float* __restrict__ Wl,
                           const float* __restrict__ Wr,
                           float* __restrict__ xl, float* __restrict__ xr) {
    __shared__ float hs[16 * 64];
    int mtile = blockIdx.x;
    int tid = threadIdx.x;
    const float* src = h1 + (size_t)mtile * 16 * 64;

#ifdef USE_TDM
    if (tid < 32) {
        tdm_load_1d((unsigned)(size_t)(void*)hs, src, 16 * 64);
        __builtin_amdgcn_s_wait_tensorcnt(0);
    }
    __syncthreads();
#else
    for (int i = tid; i < 16 * 64; i += 128) hs[i] = src[i];
    __syncthreads();
#endif

    int wave = tid >> 5;
    int lane = tid & 31;
    int wsel = wave >> 1;
    int ntile = wave & 1;
    const float* W = wsel ? Wr : Wl;
    float* Y = wsel ? xr : xl;

    int r = lane & 15;
    int half = lane >> 4;
    int coln = ntile * 16 + r;

    v8f c = {};
    #pragma unroll
    for (int kt = 0; kt < 2; kt++) {
        v16h a, b;
        #pragma unroll
        for (int p = 0; p < 8; p++) {
            int k0 = kt * 32 + ((p >> 2) << 4) + half * 8 + ((p & 3) << 1);
            a[2 * p]     = (_Float16)hs[r * 64 + k0];
            a[2 * p + 1] = (_Float16)hs[r * 64 + k0 + 1];
        }
        #pragma unroll
        for (int v = 0; v < 8; v++) {
            int kk = kt * 32 + ((v >> 2) << 4) + half * 8 + ((v & 3) << 1);
            b[2 * v]     = (_Float16)W[(size_t)kk * 32 + coln];
            b[2 * v + 1] = (_Float16)W[(size_t)(kk + 1) * 32 + coln];
        }
        c = __builtin_amdgcn_wmma_f32_16x16x32_f16(false, a, false, b,
                                                   (short)0, c, false, false);
    }
    #pragma unroll
    for (int v = 0; v < 8; v++) {
        Y[(size_t)(mtile * 16 + v + 8 * half) * 32 + coln] = c[v];
    }
}

// ---------- layer-1 edge pass A: logits + segment max (wave per edge) -------
__global__ void edge_logit_max1(const int* __restrict__ esrc, const int* __restrict__ edst,
                                const float* __restrict__ xl, const float* __restrict__ xr,
                                const float* __restrict__ att,
                                float* __restrict__ logit, unsigned* __restrict__ mx) {
    int e = blockIdx.x * 8 + (threadIdx.x >> 5);
    int lane = threadIdx.x & 31;
    if (e >= ET) return;
    int s = (e < EE) ? esrc[e] : (e - EE);
    int d = (e < EE) ? edst[e] : (e - EE);
    const float* pl = xl + (size_t)s * 64;
    const float* pr = xr + (size_t)d * 64;

    float v0 = pl[lane] + pr[lane];
    v0 = ((v0 > 0.0f) ? v0 : 0.2f * v0) * att[lane];
    float v1 = pl[32 + lane] + pr[32 + lane];
    v1 = ((v1 > 0.0f) ? v1 : 0.2f * v1) * att[32 + lane];

    v0 = wave_sum(v0);
    v1 = wave_sum(v1);
    if (lane == 0) {
        logit[(size_t)e * 2]     = v0;
        logit[(size_t)e * 2 + 1] = v1;
        atomicMax(&mx[(size_t)d * 2],     fenc(v0));
        atomicMax(&mx[(size_t)d * 2 + 1], fenc(v1));
    }
}

// ---------- layer-1 edge pass B: exp, denom, weighted agg (wave per edge) ---
__global__ void edge_acc1(const int* __restrict__ esrc, const int* __restrict__ edst,
                          const float* __restrict__ xl, const float* __restrict__ logit,
                          const unsigned* __restrict__ mx,
                          float* __restrict__ den, float* __restrict__ acc) {
    int e = blockIdx.x * 8 + (threadIdx.x >> 5);
    int lane = threadIdx.x & 31;
    if (e >= ET) return;
    int s = (e < EE) ? esrc[e] : (e - EE);
    int d = (e < EE) ? edst[e] : (e - EE);
    const float* pl = xl + (size_t)s * 64;

    float p0 = __expf(logit[(size_t)e * 2]     - fdec(mx[(size_t)d * 2]));
    float p1 = __expf(logit[(size_t)e * 2 + 1] - fdec(mx[(size_t)d * 2 + 1]));
    if (lane == 0) {
        atomicAdd(&den[(size_t)d * 2],     p0);
        atomicAdd(&den[(size_t)d * 2 + 1], p1);
    }
    atomicAdd(&acc[(size_t)d * 64 + lane],      p0 * pl[lane]);
    atomicAdd(&acc[(size_t)d * 64 + 32 + lane], p1 * pl[32 + lane]);
}

// ---------- layer-1 node finalize: bias + BN + ELU --------------------------
__global__ void node_fin1(const float* __restrict__ acc, const float* __restrict__ den,
                          const float* __restrict__ b1, const float* __restrict__ g1,
                          const float* __restrict__ be1, const float* __restrict__ m1,
                          const float* __restrict__ v1, float* __restrict__ h1) {
    int i = blockIdx.x * blockDim.x + threadIdx.x;
    if (i >= NN * 64) return;
    int n = i >> 6;
    int j = i & 63;
    float h = acc[i] / fmaxf(den[(size_t)n * 2 + (j >> 5)], 1e-16f) + b1[j];
    float bnv = (h - m1[j]) * g1[j] * rsqrtf(v1[j] + 1e-5f) + be1[j];
    h1[i] = (bnv > 0.0f) ? bnv : (__expf(bnv) - 1.0f);
}

// ---------- layer-2 edge passes (1 head, 32 ch; wave per edge) --------------
__global__ void edge_logit_max2(const int* __restrict__ esrc, const int* __restrict__ edst,
                                const float* __restrict__ xl, const float* __restrict__ xr,
                                const float* __restrict__ att,
                                float* __restrict__ logit, unsigned* __restrict__ mx) {
    int e = blockIdx.x * 8 + (threadIdx.x >> 5);
    int lane = threadIdx.x & 31;
    if (e >= ET) return;
    int s = (e < EE) ? esrc[e] : (e - EE);
    int d = (e < EE) ? edst[e] : (e - EE);
    float v = xl[(size_t)s * 32 + lane] + xr[(size_t)d * 32 + lane];
    v = ((v > 0.0f) ? v : 0.2f * v) * att[lane];
    v = wave_sum(v);
    if (lane == 0) {
        logit[e] = v;
        atomicMax(&mx[d], fenc(v));
    }
}

__global__ void edge_acc2(const int* __restrict__ esrc, const int* __restrict__ edst,
                          const float* __restrict__ xl, const float* __restrict__ logit,
                          const unsigned* __restrict__ mx,
                          float* __restrict__ den, float* __restrict__ acc) {
    int e = blockIdx.x * 8 + (threadIdx.x >> 5);
    int lane = threadIdx.x & 31;
    if (e >= ET) return;
    int s = (e < EE) ? esrc[e] : (e - EE);
    int d = (e < EE) ? edst[e] : (e - EE);
    float p = __expf(logit[e] - fdec(mx[d]));
    if (lane == 0) atomicAdd(&den[d], p);
    atomicAdd(&acc[(size_t)d * 32 + lane], p * xl[(size_t)s * 32 + lane]);
}

__global__ void node_fin2(const float* __restrict__ acc, const float* __restrict__ den,
                          const float* __restrict__ b2, const float* __restrict__ g2,
                          const float* __restrict__ be2, const float* __restrict__ m2,
                          const float* __restrict__ v2, float* __restrict__ h2) {
    int i = blockIdx.x * blockDim.x + threadIdx.x;
    if (i >= NN * 32) return;
    int n = i >> 5;
    int j = i & 31;
    float h = acc[i] / fmaxf(den[n], 1e-16f) + b2[j];
    float bnv = (h - m2[j]) * g2[j] * rsqrtf(v2[j] + 1e-5f) + be2[j];
    h2[i] = (bnv > 0.0f) ? bnv : (__expf(bnv) - 1.0f);
}

// ---------- gate MLP + segment max over batch -------------------------------
__global__ void gate_k(const float* __restrict__ h2, const int* __restrict__ batch,
                       const float* __restrict__ gw1, const float* __restrict__ gb1,
                       const float* __restrict__ gw2, const float* __restrict__ gb2,
                       float* __restrict__ gl, unsigned* __restrict__ gmx) {
    int n = blockIdx.x * blockDim.x + threadIdx.x;
    if (n >= NN) return;
    const float* hp = h2 + (size_t)n * 32;
    float o = gb2[0];
    #pragma unroll
    for (int k = 0; k < 16; k++) {
        float t = gb1[k];
        #pragma unroll
        for (int j = 0; j < 32; j++) t += hp[j] * gw1[j * 16 + k];
        t = (t > 0.0f) ? t : 0.0f;
        o += t * gw2[k];
    }
    gl[n] = o;
    atomicMax(&gmx[batch[n]], fenc(o));
}

// ---------- pooled accumulation (lane per channel) --------------------------
__global__ void pool_k(const float* __restrict__ h2, const float* __restrict__ x,
                       const int* __restrict__ batch, const float* __restrict__ gl,
                       const unsigned* __restrict__ gmx,
                       float* __restrict__ gden, float* __restrict__ pooled,
                       float* __restrict__ counts, float* __restrict__ rootc) {
    int i = blockIdx.x * blockDim.x + threadIdx.x;
    if (i >= NN * 32) return;
    int n = i >> 5;
    int j = i & 31;
    int b = batch[n];
    float p = __expf(gl[n] - fdec(gmx[b]));
    atomicAdd(&pooled[(size_t)b * 32 + j], p * h2[i]);
    if (j == 0) {
        atomicAdd(&gden[b], p);
        atomicAdd(&counts[b], 1.0f);
        atomicAdd(&rootc[b], x[(size_t)n * 26 + 24]);
    }
}

// ---------- final per-graph MLP ---------------------------------------------
__global__ void final_k(const float* __restrict__ pooled, const float* __restrict__ gden,
                        const float* __restrict__ counts, const float* __restrict__ rootc,
                        const float* __restrict__ hw1, const float* __restrict__ hb1,
                        const float* __restrict__ hw2, const float* __restrict__ hb2,
                        float* __restrict__ out) {
    int g = blockIdx.x * blockDim.x + threadIdx.x;
    if (g >= GG) return;
    float comb[33];
    float dn = fmaxf(gden[g], 1e-16f);
    #pragma unroll
    for (int j = 0; j < 32; j++) comb[j] = pooled[(size_t)g * 32 + j] / dn;
    comb[32] = rootc[g] / fmaxf(counts[g], 1.0f);
    float o = hb2[0];
    #pragma unroll
    for (int k = 0; k < 16; k++) {
        float t = hb1[k];
        #pragma unroll
        for (int j = 0; j < 33; j++) t += comb[j] * hw1[j * 16 + k];
        t = (t > 0.0f) ? t : 0.0f;
        o += t * hw2[k];
    }
    out[g] = o;
}

extern "C" void kernel_launch(void* const* d_in, const int* in_sizes, int n_in,
                              void* d_out, int out_size, void* d_ws, size_t ws_size,
                              hipStream_t stream) {
    const float* x    = (const float*)d_in[0];
    const int*   esrc = (const int*)d_in[1];
    const int*   edst = (const int*)d_in[2];
    const int*   batch= (const int*)d_in[3];
    const float* Wl1  = (const float*)d_in[4];
    const float* Wr1  = (const float*)d_in[5];
    const float* att1 = (const float*)d_in[6];
    const float* b1   = (const float*)d_in[7];
    const float* g1   = (const float*)d_in[8];
    const float* be1  = (const float*)d_in[9];
    const float* m1   = (const float*)d_in[10];
    const float* v1   = (const float*)d_in[11];
    const float* Wl2  = (const float*)d_in[12];
    const float* Wr2  = (const float*)d_in[13];
    const float* att2 = (const float*)d_in[14];
    const float* b2   = (const float*)d_in[15];
    const float* g2   = (const float*)d_in[16];
    const float* be2  = (const float*)d_in[17];
    const float* m2   = (const float*)d_in[18];
    const float* v2   = (const float*)d_in[19];
    const float* gw1  = (const float*)d_in[20];
    const float* gb1  = (const float*)d_in[21];
    const float* gw2  = (const float*)d_in[22];
    const float* gb2  = (const float*)d_in[23];
    const float* hw1  = (const float*)d_in[24];
    const float* hb1  = (const float*)d_in[25];
    const float* hw2  = (const float*)d_in[26];
    const float* hb2  = (const float*)d_in[27];
    float* out = (float*)d_out;

    // -------- workspace layout (floats) --------
    float* base = (float*)d_ws;
    size_t off = 0;
    auto alloc = [&](size_t n) { float* p = base + off; off += n; return p; };

    // no-init region
    float* xl1    = alloc((size_t)NN * 64);
    float* xr1    = alloc((size_t)NN * 64);
    float* h1     = alloc((size_t)NN * 64);
    float* xl2    = alloc((size_t)NN * 32);
    float* xr2    = alloc((size_t)NN * 32);
    float* h2     = alloc((size_t)NN * 32);
    float* logit1 = alloc((size_t)ET * 2);
    float* logit2 = alloc((size_t)ET);
    float* gl     = alloc((size_t)NN);

    // zero-init region (contiguous)
    size_t zbeg = off;
    unsigned* mx1   = (unsigned*)alloc((size_t)NN * 2);
    float*    den1  = alloc((size_t)NN * 2);
    float*    acc1  = alloc((size_t)NN * 64);
    unsigned* mx2   = (unsigned*)alloc((size_t)NN);
    float*    den2  = alloc((size_t)NN);
    float*    acc2  = alloc((size_t)NN * 32);
    unsigned* gmx   = (unsigned*)alloc((size_t)GG);
    float*    gden  = alloc((size_t)GG);
    float*    pooled= alloc((size_t)GG * 32);
    float*    counts= alloc((size_t)GG);
    float*    rootc = alloc((size_t)GG);
    size_t zcount = off - zbeg;

    const int eblocks = (ET + 7) / 8;   // wave-per-edge, 8 waves per block

    zero_ws<<<4096, 256, 0, stream>>>(base + zbeg, zcount);

    // layer 1
    transform1<<<NN / 16, 256, 0, stream>>>(x, Wl1, Wr1, xl1, xr1);
    edge_logit_max1<<<eblocks, 256, 0, stream>>>(esrc, edst, xl1, xr1, att1, logit1, mx1);
    edge_acc1<<<eblocks, 256, 0, stream>>>(esrc, edst, xl1, logit1, mx1, den1, acc1);
    node_fin1<<<(NN * 64 + 255) / 256, 256, 0, stream>>>(acc1, den1, b1, g1, be1, m1, v1, h1);

    // layer 2
    transform2<<<NN / 16, 128, 0, stream>>>(h1, Wl2, Wr2, xl2, xr2);
    edge_logit_max2<<<eblocks, 256, 0, stream>>>(esrc, edst, xl2, xr2, att2, logit2, mx2);
    edge_acc2<<<eblocks, 256, 0, stream>>>(esrc, edst, xl2, logit2, mx2, den2, acc2);
    node_fin2<<<(NN * 32 + 255) / 256, 256, 0, stream>>>(acc2, den2, b2, g2, be2, m2, v2, h2);

    // pooling + readout
    gate_k<<<(NN + 255) / 256, 256, 0, stream>>>(h2, batch, gw1, gb1, gw2, gb2, gl, gmx);
    pool_k<<<(NN * 32 + 255) / 256, 256, 0, stream>>>(h2, x, batch, gl, gmx, gden, pooled, counts, rootc);
    final_k<<<(GG + 255) / 256, 256, 0, stream>>>(pooled, gden, counts, rootc, hw1, hb1, hw2, hb2, out);
}
